// _CRU_59390807769096
// MI455X (gfx1250) — matrix operand order repaired and checked by reading, not compiled, not run on hardware
//
#include <hip/hip_runtime.h>

// ---------------------------------------------------------------------------
// CRU (3-component GRU) for MI455X (gfx1250), wave32 + WMMA bf16.
//   B=256, T=128, IN=128, H=256, 3H=768, OUT=128
// v3:
//   - persistent recurrence kernel (48 co-resident blocks), software grid
//     barrier per step; Wh tiles LDS-resident for all 128 steps.
//   - B operands staged in LDS via global_load_async_to_lds_b128 (builtin
//     confirmed present; exact v4i* AS(1)/AS(3) parameter types).
// ---------------------------------------------------------------------------

typedef __attribute__((ext_vector_type(16))) __bf16        v16bf;
typedef __attribute__((ext_vector_type(8)))  float         v8f;
typedef __attribute__((ext_vector_type(8)))  unsigned int  v8u;
typedef __attribute__((ext_vector_type(4)))  unsigned int  v4u;
typedef __attribute__((ext_vector_type(4)))  int           v4i;

#define DEV static __device__ __forceinline__

#if __has_builtin(__builtin_amdgcn_global_load_async_to_lds_b128) && \
    __has_builtin(__builtin_amdgcn_s_wait_asynccnt)
#define HAVE_ASYNC_LDS 1
#endif

DEV unsigned short f2bf(float f) {            // round-to-nearest-even f32->bf16
    unsigned int u = __float_as_uint(f);
    u += 0x7FFFu + ((u >> 16) & 1u);
    return (unsigned short)(u >> 16);
}
DEV float bf2f(unsigned short h) { return __uint_as_float(((unsigned int)h) << 16); }

// B fragment (K x 16, bf16): lane = N%16 + 16*(K>=16); 16 contiguous bf16/lane.
DEV v16bf load_b_frag(const unsigned short* p) {
    v8u r = *(const v8u*)p;                   // 32 contiguous bytes
    return __builtin_bit_cast(v16bf, r);
}
// A fragment (16 x 32, bf16): lane lo holds K {0..15}, hi {16..31}.
// Caller passes p already offset by (hi ? 16 : 0); chunks at +0 and +8.
DEV v16bf load_a_frag(const unsigned short* p) {
    v4u lo = *(const v4u*)p;
    v4u hi = *(const v4u*)(p + 8);
    v8u r;
    r[0]=lo[0]; r[1]=lo[1]; r[2]=lo[2]; r[3]=lo[3];
    r[4]=hi[0]; r[5]=hi[1]; r[6]=hi[2]; r[7]=hi[3];
    return __builtin_bit_cast(v16bf, r);
}

// copy 16 bytes global -> LDS (async path if available)
DEV void lds_fill16(unsigned short* dst, const unsigned short* src) {
#ifdef HAVE_ASYNC_LDS
    __builtin_amdgcn_global_load_async_to_lds_b128(
        (__attribute__((address_space(1))) v4i*)src,
        (__attribute__((address_space(3))) v4i*)dst, 0, 0);
#else
    *(v4u*)dst = *(const v4u*)src;
#endif
}
DEV void lds_fill_wait() {
#ifdef HAVE_ASYNC_LDS
    __builtin_amdgcn_s_wait_asynccnt(0);
#endif
}

DEV float sigmoid_f(float x) { return 1.0f / (1.0f + __expf(-x)); }
DEV float tanh_f(float x) {
    x = fminf(fmaxf(x, -20.0f), 20.0f);
    float e2 = __expf(2.0f * x);
    return (e2 - 1.0f) / (e2 + 1.0f);
}

// software grid barrier (all blocks co-resident by construction: 48 blocks)
DEV void grid_barrier(unsigned int* cnt, unsigned int* gen, unsigned int nblk) {
    __syncthreads();
    if (threadIdx.x == 0) {
        __threadfence();
        unsigned int g = __hip_atomic_load(gen, __ATOMIC_RELAXED, __HIP_MEMORY_SCOPE_AGENT);
        unsigned int a = __hip_atomic_fetch_add(cnt, 1u, __ATOMIC_ACQ_REL, __HIP_MEMORY_SCOPE_AGENT);
        if (a + 1u == nblk) {
            __hip_atomic_store(cnt, 0u, __ATOMIC_RELAXED, __HIP_MEMORY_SCOPE_AGENT);
            __hip_atomic_fetch_add(gen, 1u, __ATOMIC_RELEASE, __HIP_MEMORY_SCOPE_AGENT);
        } else {
            while (__hip_atomic_load(gen, __ATOMIC_ACQUIRE, __HIP_MEMORY_SCOPE_AGENT) == g)
                __builtin_amdgcn_s_sleep(2);
        }
        __threadfence();
    }
    __syncthreads();
}

// ------------------------------- prep kernels -------------------------------

__global__ void k_f32_to_bf16(const float* __restrict__ src,
                              unsigned short* __restrict__ dst, int n) {
    int i = blockIdx.x * 256 + threadIdx.x;
    if (i < n) dst[i] = f2bf(src[i]);
}

// dst[c*rows + r] = bf16(src[r*cols + c])  (N-major transpose for B operands)
__global__ void k_transpose_bf16(const float* __restrict__ src,
                                 unsigned short* __restrict__ dst,
                                 int rows, int cols) {
    int i = blockIdx.x * 256 + threadIdx.x;
    if (i >= rows * cols) return;
    int r = i / cols, c = i % cols;
    dst[(long)c * rows + r] = f2bf(src[i]);
}

__global__ void k_bar_init(unsigned int* p) {
    if (threadIdx.x == 0) { p[0] = 0u; p[1] = 0u; }
}

// --------------------------- Gx = X @ Wx + bx -------------------------------
// M = B*T = 32768 (Gx row m = t*256 + b), N = 768, K = 128.
// Block: 128 rows x 64 cols (8 waves, 16x64 each); WxT tile (16 KB) in LDS.
// Grid: 3072 blocks x 256 threads.
__global__ void k_gemm_gx(const unsigned short* __restrict__ xbf,   // (B*T,128) row=b*128+t
                          const unsigned short* __restrict__ WxT,   // (768,128) N-major
                          const float* __restrict__ bx,
                          unsigned short* __restrict__ Gx) {        // (t*256+b, 768)
    __shared__ __attribute__((aligned(32))) unsigned short lds_wx[64 * 128]; // 16 KB

    const int lane = threadIdx.x & 31;
    const int wid  = threadIdx.x >> 5;
    const int m0   = (blockIdx.x / 12) * 128 + wid * 16;
    const int n0   = (blockIdx.x % 12) * 64;
    const int lh   = lane & 15;
    const int hi   = lane >> 4;

    // stage B tile: WxT rows [n0, n0+64) -> contiguous 8192 bf16
    for (int i = threadIdx.x; i < 1024; i += 256)
        lds_fill16(lds_wx + i * 8, WxT + (long)n0 * 128 + i * 8);
    lds_fill_wait();
    __syncthreads();

    // per-lane A row: Gx row m -> x row  (m = t*256+b ; xrow = b*128+t)
    const int m = m0 + lh;
    const long xrow = (long)(m & 255) * 128 + (m >> 8);
    const unsigned short* ap = xbf + xrow * 128 + (hi ? 16 : 0);

    v8f acc[4] = {};
    #pragma unroll
    for (int kk = 0; kk < 128; kk += 32) {
        v16bf a = load_a_frag(ap + kk);
        const int kB = kk + (hi ? 16 : 0);
        #pragma unroll
        for (int c = 0; c < 4; ++c) {
            v16bf bfrag = load_b_frag(lds_wx + (16 * c + lh) * 128 + kB);
            acc[c] = __builtin_amdgcn_wmma_f32_16x16x32_bf16(
                false, a, false, bfrag, (short)0, acc[c], false, false);
        }
    }
    #pragma unroll
    for (int c = 0; c < 4; ++c) {
        const int n = n0 + 16 * c + lh;
        const float bxv = bx[n];
        #pragma unroll
        for (int r = 0; r < 8; ++r) {
            const int mrow = m0 + r + 8 * hi;
            Gx[(long)mrow * 768 + n] = f2bf(acc[c][r] + bxv);
        }
    }
}

// -------------------- persistent recurrence (all 128 steps) -----------------
// Block: 128 state rows x 32 hidden cols; wave w -> rows m0_blk + 16w.
// WhT rows {g*256+j0 .. +32} for 3 gates (48 KB) staged in LDS ONCE and
// reused for all 128 steps. Grid: 48 blocks x 256 threads (co-resident),
// software grid barrier per step. 48 v_wmma per wave per step.
__global__ void k_recurrence(unsigned short* __restrict__ h0,       // (768,256) bf16
                             unsigned short* __restrict__ h1,       // (768,256) bf16
                             const unsigned short* __restrict__ WhT,// (768,256) N-major
                             const unsigned short* __restrict__ Gx, // (T*256,768) bf16
                             const float* __restrict__ bh,
                             unsigned int* __restrict__ bar) {
    __shared__ __attribute__((aligned(32))) unsigned short lds_wh[3 * 32 * 256]; // 48 KB

    const int lane = threadIdx.x & 31;
    const int wid  = threadIdx.x >> 5;
    const int m0   = (blockIdx.x >> 3) * 128 + wid * 16;   // state row tile
    const int j0   = (blockIdx.x & 7) * 32;                // hidden col tile
    const int lh   = lane & 15;
    const int hi   = lane >> 4;

    // stage WhT rows for the 3 gate column-blocks: chunk g = rows [g*256+j0, +32)
    for (int i = threadIdx.x; i < 3072; i += 256) {
        const int chunk = i >> 10;           // 1024 x 16B per chunk
        const int off   = i & 1023;
        lds_fill16(lds_wh + chunk * 8192 + off * 8,
                   WhT + (long)(chunk * 256 + j0) * 256 + off * 8);
    }
    lds_fill_wait();
    __syncthreads();

    const float bhr = bh[j0 + lh], bhz = bh[256 + j0 + lh], bhn = bh[512 + j0 + lh];

    for (int t = 0; t < 128; ++t) {
        const unsigned short* h_in  = (t & 1) ? h1 : h0;
        unsigned short*       h_out = (t & 1) ? h0 : h1;
        const unsigned short* gx_t  = Gx + (long)t * 256 * 768;
        __builtin_prefetch(gx_t + (long)(m0 & 255) * 768 + j0, 0, 0);

        const unsigned short* ap = h_in + (long)(m0 + lh) * 256 + (hi ? 16 : 0);
        v8f acc[3][2] = {};
        #pragma unroll
        for (int kk = 0; kk < 256; kk += 32) {
            v16bf a = load_a_frag(ap + kk);
            const int kB = kk + (hi ? 16 : 0);
            #pragma unroll
            for (int g = 0; g < 3; ++g) {
                #pragma unroll
                for (int c = 0; c < 2; ++c) {
                    v16bf bfrag = load_b_frag(lds_wh + (g * 32 + 16 * c + lh) * 256 + kB);
                    acc[g][c] = __builtin_amdgcn_wmma_f32_16x16x32_bf16(
                        false, a, false, bfrag, (short)0, acc[g][c], false, false);
                }
            }
        }
        // fused gate math + state update
        #pragma unroll
        for (int c = 0; c < 2; ++c) {
            const int j = j0 + 16 * c + lh;
            const float br = c ? bh[j] : bhr;
            const float bz = c ? bh[256 + j] : bhz;
            const float bn = c ? bh[512 + j] : bhn;
            #pragma unroll
            for (int r = 0; r < 8; ++r) {
                const int mrow = m0 + r + 8 * hi;
                const int b    = mrow & 255;
                const unsigned short* gxp = gx_t + (long)b * 768 + j;
                const float gxr = bf2f(gxp[0]);
                const float gxz = bf2f(gxp[256]);
                const float gxn = bf2f(gxp[512]);
                const float rr = sigmoid_f(gxr + acc[0][c][r] + br);
                const float zz = sigmoid_f(gxz + acc[1][c][r] + bz);
                const float nn = tanh_f(gxn + rr * (acc[2][c][r] + bn));
                const float hold = bf2f(h_in[(long)mrow * 256 + j]);
                h_out[(long)mrow * 256 + j] = f2bf(nn + zz * (hold - nn));
            }
        }
        grid_barrier(bar, bar + 1, 48u);
    }
}

// ------------------------------- head kernels -------------------------------

__global__ void k_hsum(const unsigned short* __restrict__ h,      // (768,256)
                       unsigned short* __restrict__ hsum) {       // (256,256)
    int i = blockIdx.x * 256 + threadIdx.x;
    if (i >= 256 * 256) return;
    int b = i >> 8, j = i & 255;
    float s = bf2f(h[(long)b * 256 + j])
            + bf2f(h[(long)(256 + b) * 256 + j])
            + bf2f(h[(long)(512 + b) * 256 + j]);
    hsum[i] = f2bf(s);
}

__global__ void k_feature(const unsigned short* __restrict__ h,   // (768,256)
                          float* __restrict__ feat) {             // (3,256)
    int tid = blockIdx.x * 256 + threadIdx.x;
    if (tid >= 768) return;
    int comp = tid >> 8, j = tid & 255;
    float s = 0.0f;
    for (int b = 0; b < 256; ++b)
        s += bf2f(h[(long)(comp * 256 + b) * 256 + j]);
    feat[tid] = s * (1.0f / 256.0f);
}

// out = elu(hsum @ Wf + bf): M=256, N=128, K=256. Wave tile 16x32, 8 blocks.
__global__ void k_final(const unsigned short* __restrict__ hsum,  // (256,256) bf16
                        const unsigned short* __restrict__ WfT,   // (128,256) N-major
                        const float* __restrict__ bfv,
                        float* __restrict__ out) {                // (256,128) f32
    const int lane = threadIdx.x & 31;
    const int wid  = threadIdx.x >> 5;
    const int gw   = blockIdx.x * 8 + wid;        // 0..63
    const int m0   = (gw >> 2) * 16;
    const int n0   = (gw & 3) * 32;
    const int lh   = lane & 15;
    const int hi   = lane >> 4;

    const unsigned short* ap = hsum + (long)(m0 + lh) * 256 + (hi ? 16 : 0);
    v8f acc[2] = {};
    #pragma unroll
    for (int kk = 0; kk < 256; kk += 32) {
        v16bf a = load_a_frag(ap + kk);
        const int kB = kk + (hi ? 16 : 0);
        #pragma unroll
        for (int c = 0; c < 2; ++c) {
            v16bf bfrag = load_b_frag(WfT + (long)(n0 + 16 * c + lh) * 256 + kB);
            acc[c] = __builtin_amdgcn_wmma_f32_16x16x32_bf16(
                false, a, false, bfrag, (short)0, acc[c], false, false);
        }
    }
    #pragma unroll
    for (int c = 0; c < 2; ++c) {
        const int n = n0 + 16 * c + lh;
        const float bv = bfv[n];
        #pragma unroll
        for (int r = 0; r < 8; ++r) {
            const int mrow = m0 + r + 8 * hi;
            float v = acc[c][r] + bv;
            out[(long)mrow * 128 + n] = (v > 0.0f) ? v : (__expf(v) - 1.0f);
        }
    }
}

// ------------------------------- host launch --------------------------------

extern "C" void kernel_launch(void* const* d_in, const int* in_sizes, int n_in,
                              void* d_out, int out_size, void* d_ws, size_t ws_size,
                              hipStream_t stream) {
    (void)in_sizes; (void)n_in; (void)out_size; (void)ws_size;

    const float* x   = (const float*)d_in[0];   // (256,128,128)
    const float* hid = (const float*)d_in[1];   // (1,3,256,256)
    const float* Wx  = (const float*)d_in[2];   // (128,768)
    const float* bx  = (const float*)d_in[3];   // (768,)
    const float* Wh  = (const float*)d_in[4];   // (256,768)
    const float* bh  = (const float*)d_in[5];   // (768,)
    const float* Wf  = (const float*)d_in[6];   // (256,128)
    const float* bf  = (const float*)d_in[7];   // (128,)
    float* out = (float*)d_out;                 // 32768 out + 768 feature

    char* ws = (char*)d_ws;                     // ~60.3 MB
    unsigned short* xbf  = (unsigned short*)(ws + 0);          // 32768x128
    unsigned short* WxT  = (unsigned short*)(ws + 8388608);    // 768x128
    unsigned short* WhT  = (unsigned short*)(ws + 8585216);    // 768x256
    unsigned short* WfT  = (unsigned short*)(ws + 8978432);    // 128x256
    unsigned short* h0   = (unsigned short*)(ws + 9043968);    // 768x256
    unsigned short* h1   = (unsigned short*)(ws + 9437184);    // 768x256
    unsigned short* hsum = (unsigned short*)(ws + 9830400);    // 256x256
    unsigned short* Gx   = (unsigned short*)(ws + 9961472);    // 32768x768
    unsigned int*   bar  = (unsigned int*)  (ws + 60293120);   // {counter, generation}

    // prep: bf16 conversions + N-major weight transposes + state/barrier init
    k_f32_to_bf16<<<(4194304 + 255) / 256, 256, 0, stream>>>(x, xbf, 4194304);
    k_transpose_bf16<<<(98304 + 255) / 256, 256, 0, stream>>>(Wx, WxT, 128, 768);
    k_transpose_bf16<<<(196608 + 255) / 256, 256, 0, stream>>>(Wh, WhT, 256, 768);
    k_transpose_bf16<<<(32768 + 255) / 256, 256, 0, stream>>>(Wf, WfT, 256, 128);
    k_f32_to_bf16<<<(196608 + 255) / 256, 256, 0, stream>>>(hid, h0, 196608);
    k_bar_init<<<1, 32, 0, stream>>>(bar);

    // hoisted x-projection for all timesteps
    k_gemm_gx<<<3072, 256, 0, stream>>>(xbf, WxT, bx, Gx);

    // persistent recurrence: one launch, 128 grid-barrier-separated steps
    k_recurrence<<<48, 256, 0, stream>>>(h0, h1, WhT, Gx, bh, bar);
    // t=127 writes h0 -> final state in h0

    k_hsum<<<256, 256, 0, stream>>>(h0, hsum);
    k_feature<<<3, 256, 0, stream>>>(h0, out + 32768);
    k_final<<<8, 256, 0, stream>>>(hsum, WfT, bf, out);
}
